// DecoderLayer_33715493273894
// MI455X (gfx1250) — compile-verified, matrix operand order checked by reading, not run on hardware
//
#include <hip/hip_runtime.h>
#include <cstddef>
#include <cstdint>

typedef _Float16 f16;
typedef __attribute__((ext_vector_type(16))) _Float16 v16h;
typedef __attribute__((ext_vector_type(8)))  float    v8f;

#define B_   2
#define TQ_  1024
#define TK_  8192
#define D_   1024
#define H_   16
#define DH_  64
#define L_   512
#define FFN_ 4096
#define NEGF (-3.0e38f)

#if defined(__AMDGCN__) && __has_builtin(__builtin_amdgcn_global_load_async_to_lds_b128) && __has_builtin(__builtin_amdgcn_s_wait_asynccnt)
#define USE_ASYNC_LDS 1
typedef int i32x4 __attribute__((__vector_size__(16)));
#define ASG __attribute__((address_space(1)))
#define ASL __attribute__((address_space(3)))
#else
#define USE_ASYNC_LDS 0
#endif

__device__ __forceinline__ v8f wmma16(v16h a, v16h b, v8f c) {
    return __builtin_amdgcn_wmma_f32_16x16x32_f16(false, a, false, b, (short)0, c,
                                                  false, false);
}

// Fragment load from a row whose K-slice is contiguous (2x 16B -> 2x ds_load_b128)
__device__ __forceinline__ v16h load_frag(const f16* __restrict__ p, int hf) {
    v16h x;
#pragma unroll
    for (int j = 0; j < 8; ++j) {
        x[j]     = p[hf * 8 + j];
        x[j + 8] = p[hf * 8 + 16 + j];
    }
    return x;
}

// ---------------------------------------------------------------------------
// f32 -> f16 convert (grid-stride)
// ---------------------------------------------------------------------------
__global__ __launch_bounds__(256) void cvt_h_kernel(const float* __restrict__ in,
                                                    f16* __restrict__ out, int n)
{
    for (int i = blockIdx.x * 256 + threadIdx.x; i < n; i += gridDim.x * 256)
        out[i] = (f16)in[i];
}

// ---------------------------------------------------------------------------
// W[K,N] f32 -> Wt[N,K] f16, tiled transpose
// ---------------------------------------------------------------------------
__global__ __launch_bounds__(256) void transpose_cvt_kernel(
    const float* __restrict__ W, f16* __restrict__ Wt, int K, int N)
{
    __shared__ float tile[32][33];
    const int kb = blockIdx.y * 32, nb = blockIdx.x * 32;
    const int tx = threadIdx.x & 31, ty = threadIdx.x >> 5;
    for (int i = ty; i < 32; i += 8)
        tile[i][tx] = W[(size_t)(kb + i) * N + (nb + tx)];
    __syncthreads();
    for (int i = ty; i < 32; i += 8)
        Wt[(size_t)(nb + i) * K + (kb + tx)] = (f16)tile[tx][i];
}

// ---------------------------------------------------------------------------
// WMMA GEMM (all-f16 operands): Y = act(X[M,K] @ Wt[N,K]^T + bias)
// block 256 = 8 waves; block tile 128x128; BK=32; wave tile 32x64 (8 wmma/step)
// Async path double-buffers LDS: DMA of tile k+1 overlaps WMMA of tile k.
// HOUT==0 -> f32 out, HOUT==1 -> f16 out
// ---------------------------------------------------------------------------
template <int ACT, int HOUT>
__global__ __launch_bounds__(256) void gemm_h_kernel(
    const f16* __restrict__ X, const f16* __restrict__ Wt,
    const float* __restrict__ bias, float* __restrict__ Yf,
    f16* __restrict__ Yh, int M, int N, int K)
{
    const int tid = threadIdx.x, lane = tid & 31, wv = tid >> 5;
    const int hf = lane >> 4, r = lane & 15;
    const int m0 = blockIdx.y * 128, n0 = blockIdx.x * 128;
    const int wm = (wv & 3) * 32, wn = (wv >> 2) * 64;

    v8f acc[2][4] = {};

#if USE_ASYNC_LDS
    __shared__ __align__(16) f16 As[2][128][40];
    __shared__ __align__(16) f16 Bs[2][128][40];

    // prefetch tile 0
    for (int v = tid; v < 512; v += 256) {
        const int row = v >> 2, cv = (v & 3) * 8;
        __builtin_amdgcn_global_load_async_to_lds_b128(
            (ASG i32x4*)(X + (size_t)(m0 + row) * K + cv),
            (ASL i32x4*)&As[0][row][cv], 0, 0);
        __builtin_amdgcn_global_load_async_to_lds_b128(
            (ASG i32x4*)(Wt + (size_t)(n0 + row) * K + cv),
            (ASL i32x4*)&Bs[0][row][cv], 0, 0);
    }

    int cur = 0;
    for (int k0 = 0; k0 < K; k0 += 32) {
        __builtin_amdgcn_s_wait_asynccnt(0);   // tile k0 resident
        __syncthreads();                       // all waves done with buffer cur^1
        if (k0 + 32 < K) {                     // kick DMA for next tile
            const int kn = k0 + 32, nxt = cur ^ 1;
            for (int v = tid; v < 512; v += 256) {
                const int row = v >> 2, cv = (v & 3) * 8;
                __builtin_amdgcn_global_load_async_to_lds_b128(
                    (ASG i32x4*)(X + (size_t)(m0 + row) * K + kn + cv),
                    (ASL i32x4*)&As[nxt][row][cv], 0, 0);
                __builtin_amdgcn_global_load_async_to_lds_b128(
                    (ASG i32x4*)(Wt + (size_t)(n0 + row) * K + kn + cv),
                    (ASL i32x4*)&Bs[nxt][row][cv], 0, 0);
            }
        }
        v16h a[2], b[4];
#pragma unroll
        for (int mi = 0; mi < 2; ++mi) a[mi] = load_frag(&As[cur][wm + mi * 16 + r][0], hf);
#pragma unroll
        for (int t = 0; t < 4; ++t)    b[t]  = load_frag(&Bs[cur][wn + t * 16 + r][0], hf);
#pragma unroll
        for (int mi = 0; mi < 2; ++mi)
#pragma unroll
            for (int t = 0; t < 4; ++t)
                acc[mi][t] = wmma16(a[mi], b[t], acc[mi][t]);
        cur ^= 1;
    }
#else
    __shared__ __align__(16) f16 As[128][40];
    __shared__ __align__(16) f16 Bs[128][40];

    for (int k0 = 0; k0 < K; k0 += 32) {
        __syncthreads();
        for (int v = tid; v < 512; v += 256) {
            const int row = v >> 2, cv = (v & 3) * 8;
            *((uint4*)&As[row][cv]) = *((const uint4*)(X + (size_t)(m0 + row) * K + k0 + cv));
            *((uint4*)&Bs[row][cv]) = *((const uint4*)(Wt + (size_t)(n0 + row) * K + k0 + cv));
        }
        __syncthreads();
        v16h a[2], b[4];
#pragma unroll
        for (int mi = 0; mi < 2; ++mi) a[mi] = load_frag(&As[wm + mi * 16 + r][0], hf);
#pragma unroll
        for (int t = 0; t < 4; ++t)    b[t]  = load_frag(&Bs[wn + t * 16 + r][0], hf);
#pragma unroll
        for (int mi = 0; mi < 2; ++mi)
#pragma unroll
            for (int t = 0; t < 4; ++t)
                acc[mi][t] = wmma16(a[mi], b[t], acc[mi][t]);
    }
#endif

#pragma unroll
    for (int mi = 0; mi < 2; ++mi)
#pragma unroll
        for (int t = 0; t < 4; ++t) {
            const int col = n0 + wn + t * 16 + r;
            const float bv = bias[col];
#pragma unroll
            for (int i = 0; i < 8; ++i) {
                const int row = m0 + wm + mi * 16 + i + hf * 8;
                float v = acc[mi][t][i] + bv;
                if (ACT) v = fmaxf(v, 0.0f);
                if (HOUT) Yh[(size_t)row * N + col] = (f16)v;
                else      Yf[(size_t)row * N + col] = v;
            }
        }
}

// ---------------------------------------------------------------------------
// Flash attention kernels: block = 128 threads (4 waves), 64 queries/block
// ---------------------------------------------------------------------------
__global__ __launch_bounds__(128) void attn_self_kernel(
    const f16* __restrict__ Qh, const f16* __restrict__ Kh,
    const f16* __restrict__ Vh, const int* __restrict__ mask,
    f16* __restrict__ Ah)
{
    __shared__ __align__(16) f16 Qs[64][72];
    __shared__ __align__(16) f16 Ks[64][72];
    __shared__ __align__(16) f16 VsT[64][72];   // [dh][key]
    __shared__ __align__(16) f16 Ps[4][16][64];

    const int tid = threadIdx.x, lane = tid & 31, wv = tid >> 5;
    const int hf = lane >> 4, r = lane & 15;
    const int qb = blockIdx.x * 64, h = blockIdx.y, b = blockIdx.z;

    for (int v = tid; v < 512; v += 128) {
        const int row = v >> 3, cv = (v & 7) * 8;
        uint4 d = *((const uint4*)(Qh + ((size_t)(b * TQ_ + qb + row)) * D_ + h * DH_ + cv));
        const f16* tq = (const f16*)&d;
#pragma unroll
        for (int j = 0; j < 8; ++j)
            Qs[row][cv + j] = (f16)((float)tq[j] * 0.125f);
    }

    v8f o[4] = {};
    float m[8], l[8];
#pragma unroll
    for (int i = 0; i < 8; ++i) { m[i] = NEGF; l[i] = 0.0f; }

    const int* mB = mask + (size_t)b * TQ_ * TQ_;
    const int ar = wv * 16 + r;

    for (int kb = 0; kb < TQ_ / 64; ++kb) {
        __syncthreads();
        for (int v = tid; v < 512; v += 128) {
            const int row = v >> 3, cv = (v & 7) * 8;
            const size_t g = ((size_t)(b * TQ_ + kb * 64 + row)) * D_ + h * DH_ + cv;
#if USE_ASYNC_LDS
            __builtin_amdgcn_global_load_async_to_lds_b128(
                (ASG i32x4*)(Kh + g), (ASL i32x4*)&Ks[row][cv], 0, 0);
#else
            *((uint4*)&Ks[row][cv]) = *((const uint4*)(Kh + g));
#endif
            uint4 dv = *((const uint4*)(Vh + g));
            const f16* tv = (const f16*)&dv;
#pragma unroll
            for (int j = 0; j < 8; ++j) VsT[cv + j][row] = tv[j];
        }
#if USE_ASYNC_LDS
        __builtin_amdgcn_s_wait_asynccnt(0);
#endif
        __syncthreads();

        // S = Q K^T
        v8f s[4] = {};
#pragma unroll
        for (int kt = 0; kt < 64; kt += 32) {
            v16h a = load_frag(&Qs[ar][kt], hf);
#pragma unroll
            for (int t = 0; t < 4; ++t) {
                v16h bf = load_frag(&Ks[t * 16 + r][kt], hf);
                s[t] = wmma16(a, bf, s[t]);
            }
        }

        // mask
#pragma unroll
        for (int t = 0; t < 4; ++t) {
            const int col = kb * 64 + t * 16 + r;
#pragma unroll
            for (int i = 0; i < 8; ++i) {
                const int qrow = qb + wv * 16 + i + hf * 8;
                if (mB[(size_t)qrow * TQ_ + col] == 0) s[t][i] = NEGF;
            }
        }

        // online softmax (rows live within one 16-lane half)
#pragma unroll
        for (int i = 0; i < 8; ++i) {
            float rm = fmaxf(fmaxf(s[0][i], s[1][i]), fmaxf(s[2][i], s[3][i]));
#pragma unroll
            for (int off = 8; off >= 1; off >>= 1)
                rm = fmaxf(rm, __shfl_xor(rm, off, 32));
            const float mn = fmaxf(m[i], rm);
            const float sc = __expf(m[i] - mn);
            float rs = 0.0f;
#pragma unroll
            for (int t = 0; t < 4; ++t) {
                float p = __expf(s[t][i] - mn);
                s[t][i] = p;
                rs += p;
            }
#pragma unroll
            for (int off = 8; off >= 1; off >>= 1)
                rs += __shfl_xor(rs, off, 32);
            l[i] = l[i] * sc + rs;
            m[i] = mn;
#pragma unroll
            for (int t = 0; t < 4; ++t) o[t][i] *= sc;
        }

#pragma unroll
        for (int t = 0; t < 4; ++t)
#pragma unroll
            for (int i = 0; i < 8; ++i)
                Ps[wv][i + hf * 8][t * 16 + r] = (f16)s[t][i];
        __syncthreads();

        // O += P V   (VsT gives contiguous B fragments)
#pragma unroll
        for (int kt = 0; kt < 64; kt += 32) {
            v16h a = load_frag(&Ps[wv][r][kt], hf);
#pragma unroll
            for (int t = 0; t < 4; ++t) {
                v16h bf = load_frag(&VsT[t * 16 + r][kt], hf);
                o[t] = wmma16(a, bf, o[t]);
            }
        }
    }

#pragma unroll
    for (int i = 0; i < 8; ++i) {
        const float inv = 1.0f / l[i];
        const int qrow = qb + wv * 16 + i + hf * 8;
#pragma unroll
        for (int t = 0; t < 4; ++t)
            Ah[((size_t)(b * TQ_ + qrow)) * D_ + h * DH_ + t * 16 + r] = (f16)(o[t][i] * inv);
    }
}

__global__ __launch_bounds__(128) void attn_cross_kernel(
    const f16* __restrict__ Qh, const f16* __restrict__ Kh,
    const f16* __restrict__ Vh, const int* __restrict__ smask,
    f16* __restrict__ Ah)
{
    __shared__ __align__(16) f16 Qs[64][72];
    __shared__ __align__(16) f16 Ks[64][72];
    __shared__ __align__(16) f16 VsT[64][72];
    __shared__ __align__(16) f16 Ps[4][16][64];

    const int tid = threadIdx.x, lane = tid & 31, wv = tid >> 5;
    const int hf = lane >> 4, r = lane & 15;
    const int qb = blockIdx.x * 64, h = blockIdx.y, b = blockIdx.z;

    for (int v = tid; v < 512; v += 128) {
        const int row = v >> 3, cv = (v & 7) * 8;
        uint4 d = *((const uint4*)(Qh + ((size_t)(b * TQ_ + qb + row)) * D_ + h * DH_ + cv));
        const f16* tq = (const f16*)&d;
#pragma unroll
        for (int j = 0; j < 8; ++j)
            Qs[row][cv + j] = (f16)((float)tq[j] * 0.125f);
    }

    v8f o[4] = {};
    float m[8], l[8];
#pragma unroll
    for (int i = 0; i < 8; ++i) { m[i] = NEGF; l[i] = 0.0f; }

    const int* mB = smask + (size_t)b * TK_;
    const int ar = wv * 16 + r;

    for (int kb = 0; kb < L_ / 64; ++kb) {
        __syncthreads();
        for (int v = tid; v < 512; v += 128) {
            const int row = v >> 3, cv = (v & 7) * 8;
            const int tok = h + 16 * (kb * 64 + row);        // HEPOS strided gather
            const size_t g = ((size_t)(b * TK_ + tok)) * D_ + h * DH_ + cv;
#if USE_ASYNC_LDS
            __builtin_amdgcn_global_load_async_to_lds_b128(
                (ASG i32x4*)(Kh + g), (ASL i32x4*)&Ks[row][cv], 0, 0);
#else
            *((uint4*)&Ks[row][cv]) = *((const uint4*)(Kh + g));
#endif
            uint4 dv = *((const uint4*)(Vh + g));
            const f16* tv = (const f16*)&dv;
#pragma unroll
            for (int j = 0; j < 8; ++j) VsT[cv + j][row] = tv[j];
        }
#if USE_ASYNC_LDS
        __builtin_amdgcn_s_wait_asynccnt(0);
#endif
        __syncthreads();

        v8f s[4] = {};
#pragma unroll
        for (int kt = 0; kt < 64; kt += 32) {
            v16h a = load_frag(&Qs[ar][kt], hf);
#pragma unroll
            for (int t = 0; t < 4; ++t) {
                v16h bf = load_frag(&Ks[t * 16 + r][kt], hf);
                s[t] = wmma16(a, bf, s[t]);
            }
        }

#pragma unroll
        for (int t = 0; t < 4; ++t) {
            const int tok = h + 16 * (kb * 64 + t * 16 + r);
            if (mB[tok] == 0) {
#pragma unroll
                for (int i = 0; i < 8; ++i) s[t][i] = NEGF;
            }
        }

#pragma unroll
        for (int i = 0; i < 8; ++i) {
            float rm = fmaxf(fmaxf(s[0][i], s[1][i]), fmaxf(s[2][i], s[3][i]));
#pragma unroll
            for (int off = 8; off >= 1; off >>= 1)
                rm = fmaxf(rm, __shfl_xor(rm, off, 32));
            const float mn = fmaxf(m[i], rm);
            const float sc = __expf(m[i] - mn);
            float rs = 0.0f;
#pragma unroll
            for (int t = 0; t < 4; ++t) {
                float p = __expf(s[t][i] - mn);
                s[t][i] = p;
                rs += p;
            }
#pragma unroll
            for (int off = 8; off >= 1; off >>= 1)
                rs += __shfl_xor(rs, off, 32);
            l[i] = l[i] * sc + rs;
            m[i] = mn;
#pragma unroll
            for (int t = 0; t < 4; ++t) o[t][i] *= sc;
        }

#pragma unroll
        for (int t = 0; t < 4; ++t)
#pragma unroll
            for (int i = 0; i < 8; ++i)
                Ps[wv][i + hf * 8][t * 16 + r] = (f16)s[t][i];
        __syncthreads();

#pragma unroll
        for (int kt = 0; kt < 64; kt += 32) {
            v16h a = load_frag(&Ps[wv][r][kt], hf);
#pragma unroll
            for (int t = 0; t < 4; ++t) {
                v16h bf = load_frag(&VsT[t * 16 + r][kt], hf);
                o[t] = wmma16(a, bf, o[t]);
            }
        }
    }

#pragma unroll
    for (int i = 0; i < 8; ++i) {
        const float inv = 1.0f / l[i];
        const int qrow = qb + wv * 16 + i + hf * 8;
#pragma unroll
        for (int t = 0; t < 4; ++t)
            Ah[((size_t)(b * TQ_ + qrow)) * D_ + h * DH_ + t * 16 + r] = (f16)(o[t][i] * inv);
    }
}

// ---------------------------------------------------------------------------
// Fused residual add + LayerNorm; optional f16 mirror for downstream GEMM
// ---------------------------------------------------------------------------
template <int WH>
__global__ __launch_bounds__(256) void add_ln_kernel(
    const float* __restrict__ A, const float* __restrict__ Bt,
    const float* __restrict__ g, const float* __restrict__ beta,
    float* __restrict__ Yf, f16* __restrict__ Yh)
{
    __shared__ float red[8];
    __shared__ float row[D_];
    const int tid = threadIdx.x, lane = tid & 31, wv = tid >> 5;
    const size_t base = (size_t)blockIdx.x * D_;

    float s = 0.0f;
    for (int i = tid; i < D_; i += 256) {
        float v = A[base + i] + Bt[base + i];
        row[i] = v;
        s += v;
    }
#pragma unroll
    for (int off = 16; off >= 1; off >>= 1) s += __shfl_xor(s, off, 32);
    if (lane == 0) red[wv] = s;
    __syncthreads();
    float tot = 0.0f;
#pragma unroll
    for (int i = 0; i < 8; ++i) tot += red[i];
    const float mean = tot * (1.0f / (float)D_);

    float s2 = 0.0f;
    for (int i = tid; i < D_; i += 256) {
        float d = row[i] - mean;
        s2 += d * d;
    }
    __syncthreads();
#pragma unroll
    for (int off = 16; off >= 1; off >>= 1) s2 += __shfl_xor(s2, off, 32);
    if (lane == 0) red[wv] = s2;
    __syncthreads();
    float tot2 = 0.0f;
#pragma unroll
    for (int i = 0; i < 8; ++i) tot2 += red[i];
    const float inv = rsqrtf(tot2 * (1.0f / (float)D_) + 1e-6f);

    for (int i = tid; i < D_; i += 256) {
        float y = g[i] * (row[i] - mean) * inv + beta[i];
        Yf[base + i] = y;
        if (WH) Yh[base + i] = (f16)y;
    }
}

// ---------------------------------------------------------------------------
// Host orchestration
// ---------------------------------------------------------------------------
static inline void launch_gemm_f32out(const f16* X, const f16* Wt, const float* bias,
                                      float* Y, int M, int N, int K, hipStream_t s)
{
    dim3 grid(N / 128, M / 128);
    gemm_h_kernel<0, 0><<<grid, 256, 0, s>>>(X, Wt, bias, Y, nullptr, M, N, K);
}
static inline void launch_gemm_f16out(const f16* X, const f16* Wt, const float* bias,
                                      f16* Y, int M, int N, int K, bool relu, hipStream_t s)
{
    dim3 grid(N / 128, M / 128);
    if (relu) gemm_h_kernel<1, 1><<<grid, 256, 0, s>>>(X, Wt, bias, nullptr, Y, M, N, K);
    else      gemm_h_kernel<0, 1><<<grid, 256, 0, s>>>(X, Wt, bias, nullptr, Y, M, N, K);
}
static inline void launch_transpose(const float* W, f16* Wt, int K, int N, hipStream_t s)
{
    transpose_cvt_kernel<<<dim3(N / 32, K / 32), 256, 0, s>>>(W, Wt, K, N);
}

extern "C" void kernel_launch(void* const* d_in, const int* in_sizes, int n_in,
                              void* d_out, int out_size, void* d_ws, size_t ws_size,
                              hipStream_t stream)
{
    const float* x        = (const float*)d_in[0];
    const float* enc      = (const float*)d_in[1];
    const int*   trg_mask = (const int*)d_in[2];
    const int*   src_mask = (const int*)d_in[3];
    const float* w_f32[10] = {
        (const float*)d_in[4],  (const float*)d_in[6],  (const float*)d_in[8],
        (const float*)d_in[10], (const float*)d_in[12], (const float*)d_in[14],
        (const float*)d_in[16], (const float*)d_in[18], (const float*)d_in[26],
        (const float*)d_in[28]
    }; // sa_wq, sa_wk, sa_wv, sa_wo, ca_wq, ca_wk, ca_wv, ca_wo, ffn_w1, ffn_w2
    const float* sa_bq = (const float*)d_in[5];  const float* sa_bk = (const float*)d_in[7];
    const float* sa_bv = (const float*)d_in[9];  const float* sa_bo = (const float*)d_in[11];
    const float* ca_bq = (const float*)d_in[13]; const float* ca_bk = (const float*)d_in[15];
    const float* ca_bv = (const float*)d_in[17]; const float* ca_bo = (const float*)d_in[19];
    const float* n1g = (const float*)d_in[20];   const float* n1b = (const float*)d_in[21];
    const float* n2g = (const float*)d_in[22];   const float* n2b = (const float*)d_in[23];
    const float* n3g = (const float*)d_in[24];   const float* n3b = (const float*)d_in[25];
    const float* ffn_b1 = (const float*)d_in[27];
    const float* ffn_b2 = (const float*)d_in[29];

    const int Mq = B_ * TQ_;    // 2048
    const int Mk = B_ * TK_;    // 16384

    // ---- workspace carve-up (byte based; all chunks 16B aligned) ----
    char* p = (char*)d_ws;
    auto a16 = [&](size_t elems) { f16* q = (f16*)p; p += elems * sizeof(f16); return q; };
    auto a32 = [&](size_t elems) { float* q = (float*)p; p += elems * sizeof(float); return q; };

    f16* xh   = a16((size_t)Mq * D_);
    f16* ench = a16((size_t)Mk * D_);
    f16* wt[10];
    for (int i = 0; i < 8; ++i) wt[i] = a16((size_t)D_ * D_);
    wt[8] = a16((size_t)D_ * FFN_);      // ffn_w1t [FFN, D]
    wt[9] = a16((size_t)FFN_ * D_);      // ffn_w2t [D, FFN]
    f16* Qh  = a16((size_t)Mq * D_);
    f16* Kh  = a16((size_t)Mk * D_);
    f16* Vh  = a16((size_t)Mk * D_);
    f16* Ah  = a16((size_t)Mq * D_);
    f16* X1h = a16((size_t)Mq * D_);
    f16* X2h = a16((size_t)Mq * D_);
    f16* Fh  = a16((size_t)Mq * FFN_);
    float* bufP  = a32((size_t)Mq * D_);
    float* bufX1 = a32((size_t)Mq * D_);
    float* bufX2 = a32((size_t)Mq * D_);

    // ---- one-time precision conversion / weight transpose ----
    cvt_h_kernel<<<1024, 256, 0, stream>>>(x, xh, Mq * D_);
    cvt_h_kernel<<<2048, 256, 0, stream>>>(enc, ench, Mk * D_);
    for (int i = 0; i < 8; ++i) launch_transpose(w_f32[i], wt[i], D_, D_, stream);
    launch_transpose(w_f32[8], wt[8], D_, FFN_, stream);
    launch_transpose(w_f32[9], wt[9], FFN_, D_, stream);

    dim3 attnGrid(TQ_ / 64, H_, B_);

    // ---- self attention ----
    launch_gemm_f16out(xh, wt[0], sa_bq, Qh, Mq, D_, D_, false, stream);
    launch_gemm_f16out(xh, wt[1], sa_bk, Kh, Mq, D_, D_, false, stream);
    launch_gemm_f16out(xh, wt[2], sa_bv, Vh, Mq, D_, D_, false, stream);
    attn_self_kernel<<<attnGrid, 128, 0, stream>>>(Qh, Kh, Vh, trg_mask, Ah);
    launch_gemm_f32out(Ah, wt[3], sa_bo, bufP, Mq, D_, D_, stream);
    add_ln_kernel<1><<<Mq, 256, 0, stream>>>(x, bufP, n1g, n1b, bufX1, X1h);

    // ---- cross attention (HEPOS strided keys) ----
    launch_gemm_f16out(X1h,  wt[4], ca_bq, Qh, Mq, D_, D_, false, stream);
    launch_gemm_f16out(ench, wt[5], ca_bk, Kh, Mk, D_, D_, false, stream);
    launch_gemm_f16out(ench, wt[6], ca_bv, Vh, Mk, D_, D_, false, stream);
    attn_cross_kernel<<<attnGrid, 128, 0, stream>>>(Qh, Kh, Vh, src_mask, Ah);
    launch_gemm_f32out(Ah, wt[7], ca_bo, bufP, Mq, D_, D_, stream);
    add_ln_kernel<1><<<Mq, 256, 0, stream>>>(bufX1, bufP, n2g, n2b, bufX2, X2h);

    // ---- FFN ----
    launch_gemm_f16out(X2h, wt[8], ffn_b1, Fh, Mq, FFN_, D_, true, stream);
    launch_gemm_f32out(Fh, wt[9], ffn_b2, bufP, Mq, D_, FFN_, stream);
    add_ln_kernel<0><<<Mq, 256, 0, stream>>>(bufX2, bufP, n3g, n3b, (float*)d_out, nullptr);
}
